// ScalarQLinear_27015344292269
// MI455X (gfx1250) — matrix operand order repaired and checked by reading, not compiled, not run on hardware
//
#include <hip/hip_runtime.h>

// CDNA5 (gfx1250) wave32 WMMA types
typedef __attribute__((ext_vector_type(16))) _Float16 v16h;
typedef __attribute__((ext_vector_type(8)))  _Float16 v8h;
typedef __attribute__((ext_vector_type(4)))  _Float16 v4h;
typedef __attribute__((ext_vector_type(8)))  float    v8f;

#define BM 128
#define BN 128
#define BK 32
#define APITCH 40   // f16 per LDS row: 32 data + 8 pad -> 80B rows (16B aligned, bank-conflict-free)

// ---------------------------------------------------------------------------
// Pass 1: per-row L2 norm, quantize to 16-level integers stored as f16 (exact)
//   scale[row] = max(||w_row||, eps) / 60
//   wq[row,k]  = clip(rint(w[row,k] * 60/||w_row||), -8, 7)
// ---------------------------------------------------------------------------
__global__ __launch_bounds__(256) void quant_rows_kernel(
    const float* __restrict__ w, _Float16* __restrict__ wq,
    float* __restrict__ scale, int K)
{
  const int row = blockIdx.x;
  const float* wr = w + (size_t)row * K;

  float ss = 0.f;
  for (int k = threadIdx.x; k < K; k += 256) {
    float v = wr[k];
    ss = fmaf(v, v, ss);
  }
  __shared__ float red[256];
  red[threadIdx.x] = ss;
  __syncthreads();
  #pragma unroll
  for (int s = 128; s > 0; s >>= 1) {
    if (threadIdx.x < s) red[threadIdx.x] += red[threadIdx.x + s];
    __syncthreads();
  }
  const float norm = fmaxf(sqrtf(red[0]), 1e-8f);
  if (threadIdx.x == 0) scale[row] = norm * (1.0f / 60.0f);
  const float inv = 60.0f / norm;

  _Float16* wqr = wq + (size_t)row * K;
  for (int k = threadIdx.x; k < K; k += 256) {
    float q = rintf(wr[k] * inv);          // round-half-even, matches jnp.round
    q = fminf(fmaxf(q, -8.0f), 7.0f);      // clip to [-Q^M/2, Q^M/2 - 1]
    wqr[k] = (_Float16)q;                  // exact (small integers)
  }
}

// ---------------------------------------------------------------------------
// Pass 2: out = (x_hi + x_lo) @ wq^T * scale + bias   via v_wmma_f32_16x16x32_f16
// 128x128 C tile, 256 threads (8 waves, 4(M) x 2(N)); each wave 32x64.
// Double-buffered LDS; B tile streamed with global_load_async_to_lds_b128,
// x tile prefetched to registers and converted (f32 -> hi/lo f16) post-WMMA.
// ---------------------------------------------------------------------------
__global__ __launch_bounds__(256) void wmma_gemm_kernel(
    const float* __restrict__ x, const _Float16* __restrict__ wq,
    const float* __restrict__ scale, const float* __restrict__ bias,
    float* __restrict__ out, int M, int N, int K)
{
  __shared__ _Float16 sAhi[2][BM * APITCH];
  __shared__ _Float16 sAlo[2][BM * APITCH];
  __shared__ _Float16 sB [2][BN * APITCH];

  const int tid   = threadIdx.x;
  const int lane  = tid & 31;
  const int wave  = tid >> 5;          // 0..7
  const int wm    = wave & 3;          // wave's M slot (0..3) -> 32 rows
  const int wn    = wave >> 2;         // wave's N slot (0..1) -> 64 cols
  const int m0    = blockIdx.y * BM;
  const int n0    = blockIdx.x * BN;
  const int lhalf = lane >> 4;         // K-half select per ISA frag layout
  const int l16   = lane & 15;

  // staging work decomposition
  const int xc4 = (tid & 7) * 4;       // x col within BK (float4 granule)
  const int xrb = tid >> 3;            // x row base 0..31 (4 rows per thread)
  const int br  = tid >> 1;            // wq row 0..127
  const int bc  = (tid & 1) * 16;      // wq col 0 / 16

  const float* xrow = x + (size_t)(m0 + xrb) * K + xc4;
  const uint64_t wq_base = (uint64_t)(uintptr_t)wq;
  const uint32_t bgo0 = (uint32_t)(((size_t)(n0 + br) * K + bc) * sizeof(_Float16));

  v8f acc[2][4] = {};

  auto stage_b_async = [&](int buf, int k0) {
    // per-lane 16B async copies from global memory into LDS
    const uint32_t go = bgo0 + (uint32_t)(k0 * 2);
    const uint32_t lo = (uint32_t)(uintptr_t)&sB[buf][br * APITCH + bc];
    asm volatile("global_load_async_to_lds_b128 %0, %1, %2"
                 :: "v"(lo), "v"(go), "s"(wq_base) : "memory");
    asm volatile("global_load_async_to_lds_b128 %0, %1, %2 offset:16"
                 :: "v"(lo), "v"(go), "s"(wq_base) : "memory");
  };
  auto stage_x_regs = [&](int k0, float4* xv) {
    #pragma unroll
    for (int rr = 0; rr < 4; rr++)
      xv[rr] = *(const float4*)(xrow + (size_t)rr * 32 * K + k0);
  };
  auto stage_x_lds = [&](int buf, const float4* xv) {
    #pragma unroll
    for (int rr = 0; rr < 4; rr++) {
      const int r = rr * 32 + xrb;
      const float4 v = xv[rr];
      v4h h, l;
      h[0] = (_Float16)v.x; h[1] = (_Float16)v.y;
      h[2] = (_Float16)v.z; h[3] = (_Float16)v.w;
      l[0] = (_Float16)(v.x - (float)h[0]);
      l[1] = (_Float16)(v.y - (float)h[1]);
      l[2] = (_Float16)(v.z - (float)h[2]);
      l[3] = (_Float16)(v.w - (float)h[3]);
      *(v4h*)&sAhi[buf][r * APITCH + xc4] = h;
      *(v4h*)&sAlo[buf][r * APITCH + xc4] = l;
    }
  };

  // ---- prologue: stage tile 0 into buffer 0 ----
  {
    float4 xv[4];
    stage_b_async(0, 0);
    stage_x_regs(0, xv);
    stage_x_lds(0, xv);
  }
  asm volatile("s_wait_asynccnt 0x0" ::: "memory");  // async LDS writes landed
  __syncthreads();

  const int KT = K / BK;
  for (int kt = 0; kt < KT; kt++) {
    const int  cur = kt & 1;
    const bool pre = (kt + 1) < KT;
    float4 xv[4];
    if (pre) {
      stage_b_async(cur ^ 1, (kt + 1) * BK);   // ASYNCcnt path, no VGPRs
      stage_x_regs((kt + 1) * BK, xv);          // LOADcnt path, lands post-WMMA
    }

    // ---- fragments per ISA VGPR layouts ----
    v16h afh[2], afl[2], bf[4];
    #pragma unroll
    for (int mt = 0; mt < 2; mt++) {
      const int m = wm * 32 + mt * 16 + l16;
      const _Float16* ph = &sAhi[cur][m * APITCH + lhalf * 8];
      v8h h0 = *(const v8h*)ph, h1 = *(const v8h*)(ph + 16);
      afh[mt] = __builtin_shufflevector(h0, h1, 0,1,2,3,4,5,6,7,8,9,10,11,12,13,14,15);
      const _Float16* pl = &sAlo[cur][m * APITCH + lhalf * 8];
      v8h l0 = *(const v8h*)pl, l1 = *(const v8h*)(pl + 16);
      afl[mt] = __builtin_shufflevector(l0, l1, 0,1,2,3,4,5,6,7,8,9,10,11,12,13,14,15);
    }
    #pragma unroll
    for (int nt = 0; nt < 4; nt++) {
      const int n = wn * 64 + nt * 16 + l16;
      const _Float16* pb = &sB[cur][n * APITCH + lhalf * 8];
      v8h b0 = *(const v8h*)pb, b1 = *(const v8h*)(pb + 16);
      bf[nt] = __builtin_shufflevector(b0, b1, 0,1,2,3,4,5,6,7,8,9,10,11,12,13,14,15);
    }

    // ---- 16 WMMAs per wave per k-step (hi + lo accumulation) ----
    #pragma unroll
    for (int mt = 0; mt < 2; mt++) {
      #pragma unroll
      for (int nt = 0; nt < 4; nt++) {
        acc[mt][nt] = __builtin_amdgcn_wmma_f32_16x16x32_f16(
            false, afh[mt], false, bf[nt], (short)0, acc[mt][nt], false, false);
        acc[mt][nt] = __builtin_amdgcn_wmma_f32_16x16x32_f16(
            false, afl[mt], false, bf[nt], (short)0, acc[mt][nt], false, false);
      }
    }

    if (pre) {
      stage_x_lds(cur ^ 1, xv);                         // convert + ds_store next tile
      asm volatile("s_wait_asynccnt 0x0" ::: "memory"); // our async B writes done
      __syncthreads();                                   // (also waits DScnt)
    }
  }

  // ---- epilogue: apply per-column scale (||w_row||/60) and bias ----
  // C layout: m = 8*(lane/16) + vgpr, n = lane%16
  #pragma unroll
  for (int nt = 0; nt < 4; nt++) {
    const int n  = n0 + wn * 64 + nt * 16 + l16;
    const float s  = scale[n];
    const float bv = bias[n];
    #pragma unroll
    for (int mt = 0; mt < 2; mt++) {
      const int mbase = m0 + wm * 32 + mt * 16 + lhalf * 8;
      #pragma unroll
      for (int r = 0; r < 8; r++) {
        out[(size_t)(mbase + r) * N + n] = fmaf(acc[mt][nt][r], s, bv);
      }
    }
  }
}

// ---------------------------------------------------------------------------
extern "C" void kernel_launch(void* const* d_in, const int* in_sizes, int n_in,
                              void* d_out, int out_size, void* d_ws, size_t ws_size,
                              hipStream_t stream) {
  const float* x    = (const float*)d_in[0];
  const float* w    = (const float*)d_in[1];
  const float* bias = (const float*)d_in[2];
  float* out        = (float*)d_out;

  const int OUT = in_sizes[2];              // 4096
  const int K   = in_sizes[1] / OUT;        // 4096
  const int M   = in_sizes[0] / K;          // 8192

  // workspace: wq f16 [OUT*K] (32 MB) then scale f32 [OUT]
  _Float16* wq    = (_Float16*)d_ws;
  float*    scale = (float*)((char*)d_ws + (size_t)OUT * K * sizeof(_Float16));

  quant_rows_kernel<<<OUT, 256, 0, stream>>>(w, wq, scale, K);

  dim3 grid(OUT / BN, M / BM);
  wmma_gemm_kernel<<<grid, 256, 0, stream>>>(x, wq, scale, bias, out, M, OUT, K);
}
